// ScaledDotProductAttention_67276367725148
// MI455X (gfx1250) — matrix operand order repaired and checked by reading, not compiled, not run on hardware
//
#include <hip/hip_runtime.h>
#include <hip/hip_bf16.h>

typedef _Float16 half_t;
typedef __attribute__((ext_vector_type(16))) _Float16 v16h;
typedef __attribute__((ext_vector_type(8)))  float    v8f;

#define SQ   2048
#define DH   64
#define ROWS 16
#define NEGINF (-1e9f)

__global__ __launch_bounds__(256) void attn_wmma_kernel(
    const float* __restrict__ q, const float* __restrict__ k,
    const float* __restrict__ v, const float* __restrict__ sph,
    const int* __restrict__ mask, float* __restrict__ out,
    float* __restrict__ pattn)
{
    __shared__ half_t qtile[ROWS][DH];          // 2 KB
    __shared__ half_t pstage[8][ROWS][32];      // 8 KB  (per-wave P tile, C->A relayout)
    __shared__ float  wm[8][ROWS];              // per-wave row max
    __shared__ float  wl[8][ROWS];              // per-wave row sumexp
    __shared__ float  rowM[ROWS];
    __shared__ float  rowInv[ROWS];
    __shared__ float  opart[8][ROWS][DH];       // 32 KB (per-wave O partials)

    const int tid   = threadIdx.x;
    const int wave  = tid >> 5;
    const int lane  = tid & 31;
    const int ln    = lane & 15;      // N / M sub-index per ISA layouts
    const int hlf   = lane >> 4;      // lane-half selects K / M group

    const int h     = blockIdx.x >> 7;      // 128 q-blocks per head
    const int qb    = blockIdx.x & 127;
    const int qbase = qb * ROWS;

    const float* qh   = q    + (size_t)h * SQ * DH;
    const float* kh   = k    + (size_t)h * SQ * DH;
    const float* vh   = v    + (size_t)h * SQ * DH;
    const float* sphh = sph  + (size_t)h * SQ * SQ;
    const int*   mh   = mask + (size_t)h * SQ * SQ;
    float*       oh   = out  + (size_t)h * SQ * DH;
    float*       ph   = pattn+ (size_t)h * SQ * SQ;

    // ---- Phase 1: Q tile (pre-scaled by 1/TEMPERATURE) -> LDS f16 ----
    for (int i = tid; i < ROWS * DH; i += 256) {
        int r = i >> 6, c = i & 63;
        qtile[r][c] = (half_t)(qh[(size_t)(qbase + r) * DH + c] * 0.125f);
    }
    __syncthreads();

    // Loop-invariant A fragments (Q, 16x32 f16 layout), K chained over D=64
    v16h aq[2];
#pragma unroll
    for (int kf = 0; kf < 2; ++kf) {
        const int koff = kf * 32;
#pragma unroll
        for (int e = 0; e < 8; ++e)
            aq[kf][e] = qtile[ln][koff + 16 * hlf + e];
#pragma unroll
        for (int e = 8; e < 16; ++e)
            aq[kf][e] = qtile[ln][koff + 16 * hlf + e];
    }

    // B fragment loader for K^T: lane = key column, 16 contiguous f32 -> f16
    auto load_bk = [&](int keyrow, int koff) -> v16h {
        const float* src = kh + (size_t)keyrow * DH + koff + 16 * hlf;
        float4 f0 = *(const float4*)(src + 0);
        float4 f1 = *(const float4*)(src + 4);
        float4 f2 = *(const float4*)(src + 8);
        float4 f3 = *(const float4*)(src + 12);
        v16h b;
        b[0]=(half_t)f0.x; b[1]=(half_t)f0.y; b[2]=(half_t)f0.z; b[3]=(half_t)f0.w;
        b[4]=(half_t)f1.x; b[5]=(half_t)f1.y; b[6]=(half_t)f1.z; b[7]=(half_t)f1.w;
        b[8]=(half_t)f2.x; b[9]=(half_t)f2.y; b[10]=(half_t)f2.z; b[11]=(half_t)f2.w;
        b[12]=(half_t)f3.x; b[13]=(half_t)f3.y; b[14]=(half_t)f3.z; b[15]=(half_t)f3.w;
        return b;
    };

    // Score tiles for one 32-key slice: s = (Q/T)K^T * sph, masked -> -1e9
    auto compute_scores = [&](int kbase, v8f& s0, v8f& s1) {
#pragma unroll
        for (int nb = 0; nb < 2; ++nb) {
            const int keyrow = kbase + nb * 16 + ln;
            v16h b0 = load_bk(keyrow, 0);
            v16h b1 = load_bk(keyrow, 32);
            v8f c = {};
            c = __builtin_amdgcn_wmma_f32_16x16x32_f16(false, aq[0], false, b0,
                                                       (short)0, c, false, false);
            c = __builtin_amdgcn_wmma_f32_16x16x32_f16(false, aq[1], false, b1,
                                                       (short)0, c, false, false);
            const int col = keyrow;  // global key index == column
            // prefetch next iteration's bias/mask lines (stride 8 waves * 32 keys)
            if (col + 256 < SQ) {
                __builtin_prefetch(&sphh[(size_t)(qbase + 8 * hlf) * SQ + col + 256], 0, 0);
                __builtin_prefetch(&mh  [(size_t)(qbase + 8 * hlf) * SQ + col + 256], 0, 0);
            }
#pragma unroll
            for (int j = 0; j < 8; ++j) {
                const int row = qbase + j + 8 * hlf;
                const size_t idx = (size_t)row * SQ + col;
                float sv = c[j] * sphh[idx];
                sv = (mh[idx] == 0) ? NEGINF : sv;
                c[j] = sv;
            }
            if (nb == 0) s0 = c; else s1 = c;
        }
    };

    // ---- Pass A: online softmax statistics (flash-style, per-wave key slices) ----
    float m_run[8], l_run[8];
#pragma unroll
    for (int j = 0; j < 8; ++j) { m_run[j] = -3.0e38f; l_run[j] = 0.0f; }

    for (int it = 0; it < 8; ++it) {
        const int kbase = (it * 8 + wave) * 32;
        v8f s0, s1;
        compute_scores(kbase, s0, s1);
#pragma unroll
        for (int j = 0; j < 8; ++j) {
            float tmax = fmaxf(s0[j], s1[j]);
#pragma unroll
            for (int off = 1; off < 16; off <<= 1)
                tmax = fmaxf(tmax, __shfl_xor(tmax, off, 16));
            const float mnew = fmaxf(m_run[j], tmax);
            float se = __expf(s0[j] - mnew) + __expf(s1[j] - mnew);
#pragma unroll
            for (int off = 1; off < 16; off <<= 1)
                se += __shfl_xor(se, off, 16);
            l_run[j] = l_run[j] * __expf(m_run[j] - mnew) + se;
            m_run[j] = mnew;
        }
    }
    if (ln == 0) {
#pragma unroll
        for (int j = 0; j < 8; ++j) {
            wm[wave][j + 8 * hlf] = m_run[j];
            wl[wave][j + 8 * hlf] = l_run[j];
        }
    }
    __syncthreads();
    if (tid < ROWS) {
        float M = -3.0e38f;
        for (int w = 0; w < 8; ++w) M = fmaxf(M, wm[w][tid]);
        float L = 0.0f;
        for (int w = 0; w < 8; ++w) L += wl[w][tid] * __expf(wm[w][tid] - M);
        rowM[tid]   = M;
        rowInv[tid] = 1.0f / L;
    }
    __syncthreads();

    // ---- Pass B: recompute scores (L2-resident), emit p_attn, PV via WMMA ----
    v8f zerov = {};
    v8f oacc[4];
#pragma unroll
    for (int nb = 0; nb < 4; ++nb) oacc[nb] = zerov;

    for (int it = 0; it < 8; ++it) {
        const int kbase = (it * 8 + wave) * 32;
        v8f s0, s1;
        compute_scores(kbase, s0, s1);
#pragma unroll
        for (int nb = 0; nb < 2; ++nb) {
            v8f& s = nb ? s1 : s0;
            const int col = kbase + nb * 16 + ln;
#pragma unroll
            for (int j = 0; j < 8; ++j) {
                const int r = j + 8 * hlf;
                const float p = __expf(s[j] - rowM[r]) * rowInv[r];
                ph[(size_t)(qbase + r) * SQ + col] = p;
                pstage[wave][r][nb * 16 + ln] = (half_t)p;
            }
        }
        __syncthreads();  // uniform trip count; publishes per-wave P staging

        // A fragment of P (16x32 f16 layout) from per-wave LDS tile
        v16h ap;
#pragma unroll
        for (int e = 0; e < 8; ++e)
            ap[e] = pstage[wave][ln][16 * hlf + e];
#pragma unroll
        for (int e = 8; e < 16; ++e)
            ap[e] = pstage[wave][ln][16 * hlf + e];

        // B = V slice (32 x 64), 4 N-blocks; accumulate O partials
#pragma unroll
        for (int nb = 0; nb < 4; ++nb) {
            v16h bv;
#pragma unroll
            for (int e = 0; e < 16; ++e)
                bv[e] = (half_t)vh[(size_t)(kbase + 16 * hlf + e) * DH + nb * 16 + ln];
            oacc[nb] = __builtin_amdgcn_wmma_f32_16x16x32_f16(false, ap, false, bv,
                                                              (short)0, oacc[nb], false, false);
        }
    }

    // ---- Reduce per-wave O partials -> out ----
#pragma unroll
    for (int nb = 0; nb < 4; ++nb)
#pragma unroll
        for (int j = 0; j < 8; ++j)
            opart[wave][j + 8 * hlf][nb * 16 + ln] = oacc[nb][j];
    __syncthreads();
    for (int i = tid; i < ROWS * DH; i += 256) {
        int r = i >> 6, c = i & 63;
        float acc = 0.0f;
#pragma unroll
        for (int w = 0; w < 8; ++w) acc += opart[w][r][c];
        oh[(size_t)(qbase + r) * DH + c] = acc;
    }
}

extern "C" void kernel_launch(void* const* d_in, const int* in_sizes, int n_in,
                              void* d_out, int out_size, void* d_ws, size_t ws_size,
                              hipStream_t stream) {
    const float* q    = (const float*)d_in[0];
    const float* k    = (const float*)d_in[1];
    const float* v    = (const float*)d_in[2];
    const float* sph  = (const float*)d_in[3];
    const int*   mask = (const int*)d_in[4];
    float* out   = (float*)d_out;                       // [1,8,2048,64]
    float* pattn = out + (size_t)8 * SQ * DH;           // [1,8,2048,2048]

    dim3 grid(8 * (SQ / ROWS));   // 1024 workgroups: head x 16-row q-block
    dim3 block(256);              // 8 wave32 waves
    attn_wmma_kernel<<<grid, block, 0, stream>>>(q, k, v, sph, mask, out, pattn);
}
